// ObjectDetector_46634754899984
// MI455X (gfx1250) — matrix operand order repaired and checked by reading, not compiled, not run on hardware
//
#include <hip/hip_runtime.h>
#include <hip/hip_bf16.h>
#include <math.h>

// ---------------------------------------------------------------------------
// Problem constants (match reference)
// ---------------------------------------------------------------------------
#define NROI   1024
#define NUMFG  80
#define NCLS   81          // NUM_FG + 1
#define SCORE_THR 0.05f
#define NMS_THR   0.3f

typedef __attribute__((ext_vector_type(16))) _Float16 v16h;
typedef __attribute__((ext_vector_type(8)))  _Float16 v8h;
typedef __attribute__((ext_vector_type(4)))  _Float16 v4h;
typedef __attribute__((ext_vector_type(8)))  float    v8f;

// ---------------------------------------------------------------------------
// f32 -> f16 conversion (vectorized 4-wide). n must be a multiple of 4.
// ---------------------------------------------------------------------------
__global__ void cvt_f16(const float* __restrict__ src,
                        _Float16* __restrict__ dst, int n) {
  const int t = (blockIdx.x * blockDim.x + threadIdx.x) * 4;
  if (t >= n) return;
  const float4 v = *(const float4*)(src + t);
  v4h o;
  o[0] = (_Float16)v.x; o[1] = (_Float16)v.y;
  o[2] = (_Float16)v.z; o[3] = (_Float16)v.w;
  *(v4h*)(dst + t) = o;
}

// ---------------------------------------------------------------------------
// Pre-swizzle a weight matrix W[K x Nreal] (f32, row-major) into WMMA
// B-fragment order (f16), zero-padded to NtilesPad*16 columns.
//
// CDNA5 B fragment (32x16 f16, wave32): lane L -> n = L&15, h = L>>4;
// the 16 halves of lane L cover k = 8*h + e (e = 0..7) and k = 16 + 8*h + (e - 8) (e = 8..15) for this k-step.
// Output layout: out[((nt*K32 + kb32)*32 + lane)*16 + e]
// so a wave reads its whole fragment as two contiguous b128 loads.
// ---------------------------------------------------------------------------
__global__ void swizzle_w(const float* __restrict__ W,
                          _Float16* __restrict__ out,
                          int K, int Nreal, int NtilesPad) {
  const int K32 = K >> 5;
  const int total = NtilesPad * K32 * 32;
  const int t = blockIdx.x * blockDim.x + threadIdx.x;
  if (t >= total) return;
  const int lane = t & 31;
  const int kb32 = (t >> 5) % K32;
  const int nt   = t / (K32 * 32);

  const int n     = nt * 16 + (lane & 15);
  const int kbase = kb32 * 32 + ((lane >> 4) << 3);
  const bool nok  = (n < Nreal);

  v8h lo, hi;
#pragma unroll
  for (int e = 0; e < 8; ++e) {
    lo[e] = nok ? (_Float16)W[(size_t)(kbase + e)     * Nreal + n] : (_Float16)0.0f;
    hi[e] = nok ? (_Float16)W[(size_t)(kbase + 16 + e) * Nreal + n] : (_Float16)0.0f;
  }
  _Float16* dst = out + ((size_t)(nt * K32 + kb32) * 32 + lane) * 16;
  *(v8h*)(dst)     = lo;
  *(v8h*)(dst + 8) = hi;
}

// ---------------------------------------------------------------------------
// Register-blocked WMMA GEMM:  Y[M x Nreal] = act( A[M x K] @ W + bias )
// Each wave computes a 32x32 output patch = 2 M-tiles x 2 N-tiles
// (4 f32 accumulators), so the k-loop issues 4 v_wmma per 8 b128 loads.
//   A: f16 row-major. Per-lane A fragment = two contiguous b128:
//      halves k = 8h..8h+7 and k = 16+8h..16+8h+7 of row (m0 + lane&15).
//   B: pre-swizzled fragment buffer (see swizzle_w) = two contiguous b128.
// 4 waves per block stacked along M.
// ---------------------------------------------------------------------------
template <bool RELU, bool OUTF16>
__global__ void wmma_gemm(const _Float16* __restrict__ Ah,
                          const _Float16* __restrict__ Bf,
                          const float* __restrict__ bias,
                          void* __restrict__ Yptr,
                          int M, int K, int Nreal) {
  const int lane = threadIdx.x;                                   // 0..31
  const int mt0  = (blockIdx.y * blockDim.y + threadIdx.y) * 2;   // M-tile pair
  const int nt0  = blockIdx.x * 2;                                // N-tile pair
  const int mrow = lane & 15;
  const int h    = lane >> 4;
  const int K32  = K >> 5;

  const _Float16* arow0 = Ah + (size_t)(mt0 * 16 + mrow) * K + (h << 3);
  const _Float16* arow1 = arow0 + (size_t)16 * K;                 // next M-tile
  const _Float16* bptr0 = Bf + ((size_t)(nt0 * K32) * 32 + lane) * 16;
  const _Float16* bptr1 = bptr0 + (size_t)K32 * 512;              // next N-tile

  v8f acc00 = {}, acc01 = {}, acc10 = {}, acc11 = {};

  for (int kb = 0; kb < K32; ++kb) {
    const v8h a0lo = *(const v8h*)(arow0);
    const v8h a0hi = *(const v8h*)(arow0 + 16);
    const v8h a1lo = *(const v8h*)(arow1);
    const v8h a1hi = *(const v8h*)(arow1 + 16);
    const v8h b0lo = *(const v8h*)(bptr0);
    const v8h b0hi = *(const v8h*)(bptr0 + 8);
    const v8h b1lo = *(const v8h*)(bptr1);
    const v8h b1hi = *(const v8h*)(bptr1 + 8);
    arow0 += 32; arow1 += 32;
    bptr0 += 512; bptr1 += 512;
    __builtin_prefetch(bptr0, 0, 3);
    __builtin_prefetch(bptr1, 0, 3);
    __builtin_prefetch(arow0, 0, 3);
    __builtin_prefetch(arow1, 0, 3);

    v16h a0, a1, b0, b1;
#pragma unroll
    for (int e = 0; e < 8; ++e) {
      a0[e] = a0lo[e]; a0[e + 8] = a0hi[e];
      a1[e] = a1lo[e]; a1[e + 8] = a1hi[e];
      b0[e] = b0lo[e]; b0[e + 8] = b0hi[e];
      b1[e] = b1lo[e]; b1[e + 8] = b1hi[e];
    }
    acc00 = __builtin_amdgcn_wmma_f32_16x16x32_f16(false, a0, false, b0,
                                                   (short)0, acc00, false, false);
    acc01 = __builtin_amdgcn_wmma_f32_16x16x32_f16(false, a0, false, b1,
                                                   (short)0, acc01, false, false);
    acc10 = __builtin_amdgcn_wmma_f32_16x16x32_f16(false, a1, false, b0,
                                                   (short)0, acc10, false, false);
    acc11 = __builtin_amdgcn_wmma_f32_16x16x32_f16(false, a1, false, b1,
                                                   (short)0, acc11, false, false);
  }

  // Epilogue: C/D layout -> m = mt*16 + r + 8h, n = nt*16 + (lane&15)
#pragma unroll
  for (int q = 0; q < 4; ++q) {
    const int mt = (q >> 1) ? (mt0 + 1) : mt0;
    const int nt = (q & 1) ? (nt0 + 1) : nt0;
    const v8f acc = (q == 0) ? acc00 : (q == 1) ? acc01 : (q == 2) ? acc10 : acc11;
    const int n = nt * 16 + mrow;
    if (n < Nreal) {
      const float bv = bias[n];
#pragma unroll
      for (int r = 0; r < 8; ++r) {
        const int m = mt * 16 + r + (h << 3);
        float val = acc[r] + bv;
        if (RELU) val = fmaxf(val, 0.0f);
        if (OUTF16)
          ((_Float16*)Yptr)[(size_t)m * Nreal + n] = (_Float16)val;
        else
          ((float*)Yptr)[(size_t)m * Nreal + n] = val;
      }
    }
  }
}

// ---------------------------------------------------------------------------
// Row softmax over 81 classes, in place. One wave per row.
// ---------------------------------------------------------------------------
__global__ void softmax81(float* __restrict__ S) {
  const int row  = blockIdx.x * blockDim.y + threadIdx.y;
  const int lane = threadIdx.x;            // 0..31
  float* r = S + row * NCLS;

  const bool h1 = (lane + 32) < NCLS;
  const bool h2 = (lane + 64) < NCLS;
  const float NEG = -__builtin_inff();
  float x0 = r[lane];
  float x1 = h1 ? r[lane + 32] : NEG;
  float x2 = h2 ? r[lane + 64] : NEG;

  float mx = fmaxf(x0, fmaxf(x1, x2));
#pragma unroll
  for (int o = 16; o > 0; o >>= 1) mx = fmaxf(mx, __shfl_xor(mx, o, 32));

  float e0 = __expf(x0 - mx);
  float e1 = h1 ? __expf(x1 - mx) : 0.0f;
  float e2 = h2 ? __expf(x2 - mx) : 0.0f;
  float sm = e0 + e1 + e2;
#pragma unroll
  for (int o = 16; o > 0; o >>= 1) sm += __shfl_xor(sm, o, 32);
  const float inv = 1.0f / sm;

  r[lane] = e0 * inv;
  if (h1) r[lane + 32] = e1 * inv;
  if (h2) r[lane + 64] = e2 * inv;
}

// ---------------------------------------------------------------------------
// bbox decode: one thread per (foreground class c, roi i)
// ---------------------------------------------------------------------------
__global__ void bbox_decode(const float* __restrict__ rois,     // (N,5)
                            const float* __restrict__ im_info,  // [H,W,scale]
                            const float* __restrict__ probs,    // (N,81)
                            const float* __restrict__ deltas,   // (N,324)
                            float* __restrict__ cls_boxes,      // (80,N,4)
                            float* __restrict__ cls_scores) {   // (80,N)
  const int t = blockIdx.x * blockDim.x + threadIdx.x;
  if (t >= NUMFG * NROI) return;
  const int c = t >> 10;        // 0..79
  const int i = t & (NROI - 1); // 0..1023
  const int cc = c + 1;

  const float* rp = rois + i * 5;
  const float x1 = rp[1], y1 = rp[2], x2 = rp[3], y2 = rp[4];
  const float wdt = x2 - x1 + 1.0f;
  const float hgt = y2 - y1 + 1.0f;
  const float ctx = x1 + 0.5f * wdt;
  const float cty = y1 + 0.5f * hgt;

  const float* dp = deltas + i * (4 * NCLS) + cc * 4;
  const float dx = dp[0] * 0.1f;
  const float dy = dp[1] * 0.1f;
  const float dw = dp[2] * 0.2f;
  const float dh = dp[3] * 0.2f;

  const float pcx = dx * wdt + ctx;
  const float pcy = dy * hgt + cty;
  const float pw  = __expf(dw) * wdt;
  const float ph  = __expf(dh) * hgt;

  const float H = im_info[0], Wd = im_info[1], sc = im_info[2];
  const float bx1 = fminf(fmaxf(pcx - 0.5f * pw, 0.0f), Wd - 1.0f) / sc;
  const float by1 = fminf(fmaxf(pcy - 0.5f * ph, 0.0f), H  - 1.0f) / sc;
  const float bx2 = fminf(fmaxf(pcx + 0.5f * pw, 0.0f), Wd - 1.0f) / sc;
  const float by2 = fminf(fmaxf(pcy + 0.5f * ph, 0.0f), H  - 1.0f) / sc;

  float* ob = cls_boxes + (size_t)(c * NROI + i) * 4;
  ob[0] = bx1; ob[1] = by1; ob[2] = bx2; ob[3] = by2;
  cls_scores[c * NROI + i] = probs[i * NCLS + cc];
}

// ---------------------------------------------------------------------------
// Per-class bitonic sort (descending score, stable via index tie-break).
// One block (512 threads) per class; 1024 elements in LDS.
// ---------------------------------------------------------------------------
__global__ void class_sort(const float* __restrict__ cls_scores,
                           const float* __restrict__ cls_boxes,
                           float* __restrict__ ss,   // (80,N)
                           float* __restrict__ sb) { // (80,N,4)
  __shared__ float sc[NROI];
  __shared__ int   si[NROI];
  const int c   = blockIdx.x;
  const int tid = threadIdx.x;

  for (int t = tid; t < NROI; t += 512) {
    sc[t] = cls_scores[c * NROI + t];
    si[t] = t;
  }
  __syncthreads();

  for (int k = 2; k <= NROI; k <<= 1) {
    for (int j = k >> 1; j > 0; j >>= 1) {
      for (int t = tid; t < NROI; t += 512) {
        const int ixj = t ^ j;
        if (ixj > t) {
          const bool up = ((t & k) == 0);  // "up" == desired (descending) order
          const float s1 = sc[t], s2 = sc[ixj];
          const int   i1 = si[t], i2 = si[ixj];
          const bool inOrder = (s1 > s2) || (s1 == s2 && i1 < i2);
          if (up ? !inOrder : inOrder) {
            sc[t] = s2; sc[ixj] = s1;
            si[t] = i2; si[ixj] = i1;
          }
        }
      }
      __syncthreads();
    }
  }

  for (int t = tid; t < NROI; t += 512) {
    ss[c * NROI + t] = sc[t];
    const int src = si[t];
    const float* bp = cls_boxes + (size_t)(c * NROI + src) * 4;
    float* op = sb + (size_t)(c * NROI + t) * 4;
    op[0] = bp[0]; op[1] = bp[1]; op[2] = bp[2]; op[3] = bp[3];
  }
}

// ---------------------------------------------------------------------------
// Per-class greedy NMS + dets emission. One block (256 threads) per class.
// keep[i] = valid[i] && !any(j<i : keep[j] && iou(i,j) > thr)
// ---------------------------------------------------------------------------
__global__ void nms_emit(const float* __restrict__ sb,  // (80,N,4) sorted
                         const float* __restrict__ ss,  // (80,N)   sorted
                         float* __restrict__ out) {     // (80,N,6)
  constexpr int NT = 256;
  __shared__ float lx1[NROI], ly1[NROI], lx2[NROI], ly2[NROI], lar[NROI];
  __shared__ unsigned char lval[NROI];
  __shared__ unsigned kept[NROI / 32];
  __shared__ unsigned wflag[NT / 32];

  const int c   = blockIdx.x;
  const int tid = threadIdx.x;

  for (int t = tid; t < NROI; t += NT) {
    const float* bp = sb + (size_t)(c * NROI + t) * 4;
    const float X1 = bp[0], Y1 = bp[1], X2 = bp[2], Y2 = bp[3];
    lx1[t] = X1; ly1[t] = Y1; lx2[t] = X2; ly2[t] = Y2;
    lar[t] = (X2 - X1 + 1.0f) * (Y2 - Y1 + 1.0f);
    lval[t] = (ss[c * NROI + t] > SCORE_THR) ? 1 : 0;
  }
  if (tid < NROI / 32) kept[tid] = 0u;
  __syncthreads();

  for (int i = 0; i < NROI; ++i) {
    const float ix1 = lx1[i], iy1 = ly1[i], ix2 = lx2[i], iy2 = ly2[i];
    const float ia  = lar[i];
    bool over = false;
    for (int j = tid; j < i; j += NT) {
      if (kept[j >> 5] & (1u << (j & 31))) {
        const float iw = fminf(ix2, lx2[j]) - fmaxf(ix1, lx1[j]) + 1.0f;
        const float ih = fminf(iy2, ly2[j]) - fmaxf(iy1, ly1[j]) + 1.0f;
        if (iw > 0.0f && ih > 0.0f) {
          const float inter = iw * ih;
          const float iou = inter / (ia + lar[j] - inter);
          over |= (iou > NMS_THR);
        }
      }
    }
    const unsigned bal = (unsigned)__ballot(over);  // wave32: low 32 bits
    if ((tid & 31) == 0) wflag[tid >> 5] = bal;
    __syncthreads();
    if (tid == 0) {
      unsigned any = 0;
#pragma unroll
      for (int w = 0; w < NT / 32; ++w) any |= wflag[w];
      if (lval[i] && !any) kept[i >> 5] |= (1u << (i & 31));
    }
    __syncthreads();
  }

  for (int t = tid; t < NROI; t += NT) {
    const bool k = (kept[t >> 5] >> (t & 31)) & 1u;
    float* o = out + (size_t)(c * NROI + t) * 6;
    if (k) {
      o[0] = lx1[t]; o[1] = ly1[t]; o[2] = lx2[t]; o[3] = ly2[t];
      o[4] = ss[c * NROI + t];
      o[5] = (float)c;
    } else {
      o[0] = 0.0f; o[1] = 0.0f; o[2] = 0.0f;
      o[3] = 0.0f; o[4] = 0.0f; o[5] = 0.0f;
    }
  }
}

// ---------------------------------------------------------------------------
// Host-side launcher
// in order: pooled_features, rois, im_info, w1,b1, w2,b2, w3,b3, w4,b4
// ---------------------------------------------------------------------------
extern "C" void kernel_launch(void* const* d_in, const int* in_sizes, int n_in,
                              void* d_out, int out_size, void* d_ws, size_t ws_size,
                              hipStream_t stream) {
  const float* X       = (const float*)d_in[0];   // (1024, 2048)
  const float* rois    = (const float*)d_in[1];   // (1, 1024, 5)
  const float* im_info = (const float*)d_in[2];   // (1, 3)
  const float* w1 = (const float*)d_in[3];
  const float* b1 = (const float*)d_in[4];
  const float* w2 = (const float*)d_in[5];
  const float* b2 = (const float*)d_in[6];
  const float* w3 = (const float*)d_in[7];
  const float* b3 = (const float*)d_in[8];
  const float* w4 = (const float*)d_in[9];
  const float* b4 = (const float*)d_in[10];
  float* out = (float*)d_out;                     // (80, 1024, 6)

  // ---- workspace carve-up (byte offsets, all 256-aligned; total ~13.6 MB)
  char* ws = (char*)d_ws;
  _Float16* Xh   = (_Float16*)(ws + 0);          // 1024*2048 f16        = 4194304
  _Float16* w1f  = (_Float16*)(ws + 4194304);    // 16*64*512 h          = 1048576
  _Float16* w3f  = (_Float16*)(ws + 5242880);    // 32*64*512 h          = 2097152
  _Float16* w2f  = (_Float16*)(ws + 7340032);    // 6*8*512 h            = 49152
  _Float16* w4f  = (_Float16*)(ws + 7389184);    // 22*16*512 h          = 360448
  _Float16* Y1h  = (_Float16*)(ws + 7749632);    // 1024*256 f16         = 524288
  _Float16* Y3h  = (_Float16*)(ws + 8273920);    // 1024*512 f16         = 1048576
  float* S          = (float*)(ws + 9322496);    // 1024*81  f32         = 331776
  float* D          = (float*)(ws + 9654272);    // 1024*324 f32         = 1327104
  float* cls_boxes  = (float*)(ws + 10981376);   // 80*1024*4 f32        = 1310720
  float* cls_scores = (float*)(ws + 12292096);   // 80*1024  f32         = 327680
  float* sbuf       = (float*)(ws + 12619776);   // 80*1024*4 f32        = 1310720
  float* ssort      = (float*)(ws + 13930496);   // 80*1024  f32         = 327680
  (void)ws_size; (void)in_sizes; (void)n_in; (void)out_size;

  // ---- 1) one-time conversions / weight swizzles (all tiny, L2-resident)
  cvt_f16<<<(1024 * 2048 / 4 + 255) / 256, 256, 0, stream>>>(X, Xh, 1024 * 2048);
  swizzle_w<<<(16 * 64 * 32 + 255) / 256, 256, 0, stream>>>(w1, w1f, 2048, 256, 16);
  swizzle_w<<<(32 * 64 * 32 + 255) / 256, 256, 0, stream>>>(w3, w3f, 2048, 512, 32);
  swizzle_w<<<( 6 *  8 * 32 + 255) / 256, 256, 0, stream>>>(w2, w2f,  256,  81,  6);
  swizzle_w<<<(22 * 16 * 32 + 255) / 256, 256, 0, stream>>>(w4, w4f,  512, 324, 22);

  // ---- 2) register-blocked WMMA GEMMs
  //      block = 4 waves; each wave owns a 32x32 patch (2 M-tiles x 2 N-tiles)
  //      grid.x = NtilesPad/2, grid.y = M / (16*2*4) = 8
  const dim3 gblk(32, 4);
  // Y1 = relu(X @ w1 + b1)   M=1024 K=2048 N=256  -> f16
  wmma_gemm<true, true><<<dim3(8, 8), gblk, 0, stream>>>(Xh, w1f, b1, Y1h, 1024, 2048, 256);
  // Y3 = relu(X @ w3 + b3)   M=1024 K=2048 N=512  -> f16
  wmma_gemm<true, true><<<dim3(16, 8), gblk, 0, stream>>>(Xh, w3f, b3, Y3h, 1024, 2048, 512);
  // S  = Y1 @ w2 + b2        M=1024 K=256  N=81   -> f32
  wmma_gemm<false, false><<<dim3(3, 8), gblk, 0, stream>>>(Y1h, w2f, b2, S, 1024, 256, 81);
  // D  = Y3 @ w4 + b4        M=1024 K=512  N=324  -> f32
  wmma_gemm<false, false><<<dim3(11, 8), gblk, 0, stream>>>(Y3h, w4f, b4, D, 1024, 512, 324);

  // ---- 3) softmax over 81 classes (in place on S)
  softmax81<<<1024 / 8, dim3(32, 8), 0, stream>>>(S);

  // ---- 4) per-(class, roi) bbox decode
  bbox_decode<<<(NUMFG * NROI + 255) / 256, 256, 0, stream>>>(
      rois, im_info, S, D, cls_boxes, cls_scores);

  // ---- 5) per-class sort (desc score, stable)
  class_sort<<<NUMFG, 512, 0, stream>>>(cls_scores, cls_boxes, ssort, sbuf);

  // ---- 6) per-class NMS + dets emission
  nms_emit<<<NUMFG, 256, 0, stream>>>(sbuf, ssort, out);
}
